// GraphRec_82806969467404
// MI455X (gfx1250) — matrix operand order repaired and checked by reading, not compiled, not run on hardware
//
#include <hip/hip_runtime.h>

typedef __attribute__((ext_vector_type(16))) _Float16 v16h;
typedef __attribute__((ext_vector_type(8)))  _Float16 v8h;
typedef __attribute__((ext_vector_type(8)))  float    v8f;

#define BB 4096
#define LL 200
#define DD 64
#define STG 72        // padded stage stride in halves (144 B -> 16B-aligned b128 LDS loads)
#define TPW 8         // tiles per wave in the branch kernel (amortizes weight packing)

// ---- WMMA wrapper (codegen-confirmed builtin, 8 args) ----
static __device__ __forceinline__ v8f wmma_f16(v16h a, v16h b, v8f c) {
  return __builtin_amdgcn_wmma_f32_16x16x32_f16(false, a, false, b, (short)0, c, false, false);
}

// Pack a K x N (row-major, f32) weight matrix into WMMA B-fragment layout in LDS:
// packed[(kt*NT+nt)*32 + lane][h] = W[32*kt + 8*(lane>=16) + h + 8*(h>=8)][16*nt + (lane&15)]
static __device__ __forceinline__ void pack_w(const float* __restrict__ W,
                                              _Float16* dst, int K, int N) {
  const int NT = N >> 4;
  const int total = (K >> 5) * NT * 32;
  for (int idx = threadIdx.x; idx < total; idx += blockDim.x) {
    const int l    = idx & 31;
    const int tile = idx >> 5;
    const int nt   = tile % NT;
    const int kt   = tile / NT;
    const int n  = (nt << 4) + (l & 15);
    const int kb = (kt << 5) + ((l >> 4) << 3);
    _Float16* o = dst + idx * 16;
#pragma unroll
    for (int h = 0; h < 16; ++h) o[h] = (_Float16)W[(kb + h + ((h >> 3) << 3)) * N + n];
  }
}

static __device__ __forceinline__ v16h wfrag(const _Float16* base, int kt, int nt, int NT, int l) {
  return *(const v16h*)(base + (((kt * NT) + nt) * 32 + l) * 16);
}

// Build one 16x32 f16 A-fragment from a dense f32 row (32 consecutive columns),
// per ISA A layout: halves 0..7 -> k=o..o+7 ; halves 8..15 -> k=o+16..o+23, o = 8*(lane>=16)
static __device__ __forceinline__ v16h row_frag(const float* __restrict__ p, int o, float s) {
  const float4 a = *(const float4*)(p + o);
  const float4 b = *(const float4*)(p + o + 4);
  const float4 c = *(const float4*)(p + o + 16);
  const float4 d = *(const float4*)(p + o + 20);
  v16h r;
  r[0]=(_Float16)(a.x*s);  r[1]=(_Float16)(a.y*s);  r[2]=(_Float16)(a.z*s);  r[3]=(_Float16)(a.w*s);
  r[4]=(_Float16)(b.x*s);  r[5]=(_Float16)(b.y*s);  r[6]=(_Float16)(b.z*s);  r[7]=(_Float16)(b.w*s);
  r[8]=(_Float16)(c.x*s);  r[9]=(_Float16)(c.y*s);  r[10]=(_Float16)(c.z*s); r[11]=(_Float16)(c.w*s);
  r[12]=(_Float16)(d.x*s); r[13]=(_Float16)(d.y*s); r[14]=(_Float16)(d.z*s); r[15]=(_Float16)(d.w*s);
  return r;
}

// A-fragment from the per-wave LDS staging tile (16 rows x STG halves)
static __device__ __forceinline__ v16h stage_frag(const _Float16* st, int l, int t) {
  const int m = l & 15;
  const int o = (l >> 4) << 3;
  const _Float16* row = st + m * STG + t * 32 + o;
  v8h lo = *(const v8h*)(row);
  v8h hi = *(const v8h*)(row + 16);
  v16h r;
#pragma unroll
  for (int i = 0; i < 8; ++i) { r[i] = lo[i]; r[8 + i] = hi[i]; }
  return r;
}

// Scatter a D-fragment (n-tile j) into the staging tile as f16
static __device__ __forceinline__ void stage_put(_Float16* st, int l, int j, v8f v) {
  const int n  = (j << 4) + (l & 15);
  const int mb = (l >> 4) << 3;
#pragma unroll
  for (int r = 0; r < 8; ++r) st[(mb + r) * STG + n] = (_Float16)v[r];
}

static __device__ __forceinline__ v8f bias_relu(v8f v, float b) {
#pragma unroll
  for (int r = 0; r < 8; ++r) v[r] = fmaxf(v[r] + b, 0.0f);
  return v;
}

// ================= fused per-(b,l) branch kernel =================
// Computes f = g_MLP([nb|er]) (f32 D-frags), a' = exp(att_MLP([f|mask*self]))*mask,
// then atomically accumulates num[b,:] += a'*f and den[b] += a'.
__global__ __launch_bounds__(256) void graphrec_branch(
    const int* __restrict__ ids, const int* __restrict__ pad,
    const float* __restrict__ self_tbl, const float* __restrict__ neigh_tbl,
    const float* __restrict__ rate_tbl,
    const float* g0w, const float* g0b, const float* g1w, const float* g1b,
    const float* g2w, const float* g2b,
    const float* a0w, const float* a0b, const float* a1w, const float* a1b,
    const float* a2w, const float* a2b,
    float* __restrict__ num, float* __restrict__ den)
{
  __shared__ __align__(32) _Float16 sW1[4 * 4 * 32 * 16];   // g  L1: 128x64
  __shared__ __align__(32) _Float16 sA1[4 * 4 * 32 * 16];   // att L1: 128x64
  __shared__ __align__(32) _Float16 sW2[2 * 2 * 32 * 16];   // g  L2: 64x32
  __shared__ __align__(32) _Float16 sA2[2 * 2 * 32 * 16];   // att L2: 64x32
  __shared__ __align__(32) _Float16 sW3[1 * 4 * 32 * 16];   // g  L3: 32x64
  __shared__ __align__(16) _Float16 sStage[8][16 * STG];    // per-wave staging
  __shared__ float sB1[64], sB3[64], sAB1[64];
  __shared__ float sB2[32], sAB2[32], sA3w[32];
  __shared__ float sAw[8][16];
  __shared__ float sA3b;

  pack_w(g0w, sW1, 128, 64);
  pack_w(a0w, sA1, 128, 64);
  pack_w(g1w, sW2, 64, 32);
  pack_w(a1w, sA2, 64, 32);
  pack_w(g2w, sW3, 32, 64);
  for (int i = threadIdx.x; i < 64; i += blockDim.x) { sB1[i]=g0b[i]; sB3[i]=g2b[i]; sAB1[i]=a0b[i]; }
  for (int i = threadIdx.x; i < 32; i += blockDim.x) { sB2[i]=g1b[i]; sAB2[i]=a1b[i]; sA3w[i]=a2w[i]; }
  if (threadIdx.x == 0) sA3b = a2b[0];
  __syncthreads();

  const int wave = threadIdx.x >> 5;
  const int l    = threadIdx.x & 31;
  _Float16* stage = &sStage[wave][0];
  const int m  = l & 15;
  const int o  = (l >> 4) << 3;
  const int mb = (l >> 4) << 3;
  const int nn = l & 15;

  // each wave owns TPW consecutive 16-row tiles -> contiguous pad reads, amortized packing
  const int tile0 = (blockIdx.x * 8 + wave) * TPW;

  for (int it = 0; it < TPW; ++it) {
    const int base = (tile0 + it) * 16;
    const int gr   = base + m;                   // this lane's row (2 lanes per row)
    const int nid  = pad[gr * 2 + 0];
    const int rid  = pad[gr * 2 + 1];
    const int bidx = gr / LL;
    const float msk = (nid > 0) ? 1.0f : 0.0f;   // padding rows: table row 0 is all-zero anyway

    // ---- X = [neigh_emb | rate_emb] A-fragments (16x128) via gathers ----
    v16h xa[4];
    xa[0] = row_frag(neigh_tbl + (size_t)nid * DD,      o, 1.0f);
    xa[1] = row_frag(neigh_tbl + (size_t)nid * DD + 32, o, 1.0f);
    xa[2] = row_frag(rate_tbl  + (size_t)rid * DD,      o, 1.0f);
    xa[3] = row_frag(rate_tbl  + (size_t)rid * DD + 32, o, 1.0f);

    // ---- g layer 1: [16,128]x[128,64] + bias + ReLU ----
#pragma unroll
    for (int j = 0; j < 4; ++j) {
      v8f acc = {0,0,0,0,0,0,0,0};
#pragma unroll
      for (int t = 0; t < 4; ++t) acc = wmma_f16(xa[t], wfrag(sW1, t, j, 4, l), acc);
      acc = bias_relu(acc, sB1[(j << 4) + nn]);
      stage_put(stage, l, j, acc);
    }
    asm volatile("" ::: "memory");
    // ---- g layer 2: [16,64]x[64,32] ----
    v16h h1a0 = stage_frag(stage, l, 0);
    v16h h1a1 = stage_frag(stage, l, 1);
#pragma unroll
    for (int j = 0; j < 2; ++j) {
      v8f acc = {0,0,0,0,0,0,0,0};
      acc = wmma_f16(h1a0, wfrag(sW2, 0, j, 2, l), acc);
      acc = wmma_f16(h1a1, wfrag(sW2, 1, j, 2, l), acc);
      acc = bias_relu(acc, sB2[(j << 4) + nn]);
      stage_put(stage, l, j, acc);
    }
    asm volatile("" ::: "memory");
    // ---- g layer 3: [16,32]x[32,64] + bias (no ReLU) -> f kept in f32 frags ----
    v16h h2a = stage_frag(stage, l, 0);
    v8f fr[4];
#pragma unroll
    for (int j = 0; j < 4; ++j) {
      v8f acc = {0,0,0,0,0,0,0,0};
      acc = wmma_f16(h2a, wfrag(sW3, 0, j, 4, l), acc);
      const float b = sB3[(j << 4) + nn];
#pragma unroll
      for (int r = 0; r < 8; ++r) acc[r] += b;
      fr[j] = acc;
      stage_put(stage, l, j, acc);               // f16 copy of f for attention input
    }
    asm volatile("" ::: "memory");

    // ---- attention layer 1 on [f | mask*self] ----
    v16h ta[4];
    ta[0] = stage_frag(stage, l, 0);
    ta[1] = stage_frag(stage, l, 1);
    const float* selfrow = self_tbl + (size_t)ids[bidx] * DD;
    ta[2] = row_frag(selfrow,      o, msk);
    ta[3] = row_frag(selfrow + 32, o, msk);
#pragma unroll
    for (int j = 0; j < 4; ++j) {
      v8f acc = {0,0,0,0,0,0,0,0};
#pragma unroll
      for (int t = 0; t < 4; ++t) acc = wmma_f16(ta[t], wfrag(sA1, t, j, 4, l), acc);
      acc = bias_relu(acc, sAB1[(j << 4) + nn]);
      stage_put(stage, l, j, acc);
    }
    asm volatile("" ::: "memory");
    // ---- attention layer 2 ----
    v16h q0 = stage_frag(stage, l, 0);
    v16h q1 = stage_frag(stage, l, 1);
#pragma unroll
    for (int j = 0; j < 2; ++j) {
      v8f acc = {0,0,0,0,0,0,0,0};
      acc = wmma_f16(q0, wfrag(sA2, 0, j, 2, l), acc);
      acc = wmma_f16(q1, wfrag(sA2, 1, j, 2, l), acc);
      acc = bias_relu(acc, sAB2[(j << 4) + nn]);
      stage_put(stage, l, j, acc);
    }
    asm volatile("" ::: "memory");
    // ---- attention layer 3: [16,32]x[32,1]; a' = exp(logit)*mask ----
    if (l < 16) {                                // lane l<16 has m==l
      float s = sA3b;
#pragma unroll
      for (int k = 0; k < 32; ++k) s += (float)stage[l * STG + k] * sA3w[k];
      const float ap = (nid > 0) ? __expf(s) : 0.0f;
      sAw[wave][l] = ap;
      atomicAdd(den + bidx, ap);
    }
    asm volatile("" ::: "memory");

    // ---- accumulate numerator: num[b,:] += a'_row * f_row ----
    // The 8 rows base+mb .. base+mb+7 span at most 2 batch indices (L=200),
    // so pre-reduce to <=2 sums per (j,lane) before hitting global atomics.
    const int b0 = (base + mb) / LL;
#pragma unroll
    for (int j = 0; j < 4; ++j) {
      float s0 = 0.0f, s1 = 0.0f;
#pragma unroll
      for (int r = 0; r < 8; ++r) {
        const int mm = mb + r;
        const float v = sAw[wave][mm] * fr[j][r];
        if ((base + mm) / LL == b0) s0 += v; else s1 += v;
      }
      atomicAdd(num + (size_t)b0 * DD + (j << 4) + nn, s0);
      if (s1 != 0.0f) atomicAdd(num + (size_t)(b0 + 1) * DD + (j << 4) + nn, s1);
    }
  }
}

// ================= aggregation: h = relu(relu(x@W0+b0)@W1+b1), x = num/(den+eps) =================
__global__ __launch_bounds__(256) void graphrec_agg(
    const float* __restrict__ num, const float* __restrict__ den,
    const float* w0, const float* b0, const float* w1, const float* b1,
    float* __restrict__ out)
{
  __shared__ __align__(32) _Float16 sG0[2 * 4 * 32 * 16];
  __shared__ __align__(32) _Float16 sG1[2 * 4 * 32 * 16];
  __shared__ __align__(16) _Float16 sStage[8][16 * STG];
  __shared__ float sBB0[64], sBB1[64];

  pack_w(w0, sG0, 64, 64);
  pack_w(w1, sG1, 64, 64);
  for (int i = threadIdx.x; i < 64; i += blockDim.x) { sBB0[i] = b0[i]; sBB1[i] = b1[i]; }
  __syncthreads();

  const int wave = threadIdx.x >> 5;
  const int l = threadIdx.x & 31;
  const int base = (blockIdx.x * 8 + wave) * 16;
  _Float16* stage = &sStage[wave][0];
  const int m = l & 15, o = (l >> 4) << 3;
  const int row = base + m;
  const float invd = 1.0f / (den[row] + 1e-10f);

  v16h xa0 = row_frag(num + (size_t)row * DD,      o, invd);
  v16h xa1 = row_frag(num + (size_t)row * DD + 32, o, invd);
#pragma unroll
  for (int j = 0; j < 4; ++j) {
    v8f acc = {0,0,0,0,0,0,0,0};
    acc = wmma_f16(xa0, wfrag(sG0, 0, j, 4, l), acc);
    acc = wmma_f16(xa1, wfrag(sG0, 1, j, 4, l), acc);
    acc = bias_relu(acc, sBB0[(j << 4) + (l & 15)]);
    stage_put(stage, l, j, acc);
  }
  asm volatile("" ::: "memory");
  v16h q0 = stage_frag(stage, l, 0);
  v16h q1 = stage_frag(stage, l, 1);
#pragma unroll
  for (int j = 0; j < 4; ++j) {
    v8f acc = {0,0,0,0,0,0,0,0};
    acc = wmma_f16(q0, wfrag(sG1, 0, j, 4, l), acc);
    acc = wmma_f16(q1, wfrag(sG1, 1, j, 4, l), acc);
    acc = bias_relu(acc, sBB1[(j << 4) + (l & 15)]);
    const int n = (j << 4) + (l & 15);
    const int mbo = (l >> 4) << 3;
#pragma unroll
    for (int r = 0; r < 8; ++r) out[(size_t)(base + mbo + r) * DD + n] = acc[r];
  }
}

// ================= rate predictor: 4x (Linear+ReLU) on [h|z] =================
__global__ __launch_bounds__(256) void graphrec_rate(
    const float* __restrict__ h, const float* __restrict__ z,
    const float* w0, const float* b0, const float* w1, const float* b1,
    const float* w2, const float* b2, const float* w3, const float* b3,
    float* __restrict__ out)
{
  __shared__ __align__(32) _Float16 sR0[4 * 4 * 32 * 16];
  __shared__ __align__(32) _Float16 sR1[2 * 4 * 32 * 16];
  __shared__ __align__(32) _Float16 sR2[2 * 4 * 32 * 16];
  __shared__ __align__(16) _Float16 sStage[8][16 * STG];
  __shared__ float sBB0[64], sBB1[64], sBB2[64], sW4[64];
  __shared__ float sB4;

  pack_w(w0, sR0, 128, 64);
  pack_w(w1, sR1, 64, 64);
  pack_w(w2, sR2, 64, 64);
  for (int i = threadIdx.x; i < 64; i += blockDim.x) { sBB0[i]=b0[i]; sBB1[i]=b1[i]; sBB2[i]=b2[i]; sW4[i]=w3[i]; }
  if (threadIdx.x == 0) sB4 = b3[0];
  __syncthreads();

  const int wave = threadIdx.x >> 5;
  const int l = threadIdx.x & 31;
  const int base = (blockIdx.x * 8 + wave) * 16;
  _Float16* stage = &sStage[wave][0];
  const int m = l & 15, o = (l >> 4) << 3;
  const int row = base + m;

  v16h xa[4];
  xa[0] = row_frag(h + (size_t)row * DD,      o, 1.0f);
  xa[1] = row_frag(h + (size_t)row * DD + 32, o, 1.0f);
  xa[2] = row_frag(z + (size_t)row * DD,      o, 1.0f);
  xa[3] = row_frag(z + (size_t)row * DD + 32, o, 1.0f);

#pragma unroll
  for (int j = 0; j < 4; ++j) {               // L1 128->64
    v8f acc = {0,0,0,0,0,0,0,0};
#pragma unroll
    for (int t = 0; t < 4; ++t) acc = wmma_f16(xa[t], wfrag(sR0, t, j, 4, l), acc);
    acc = bias_relu(acc, sBB0[(j << 4) + (l & 15)]);
    stage_put(stage, l, j, acc);
  }
  asm volatile("" ::: "memory");
  {                                           // L2 64->64
    v16h q0 = stage_frag(stage, l, 0);
    v16h q1 = stage_frag(stage, l, 1);
#pragma unroll
    for (int j = 0; j < 4; ++j) {
      v8f acc = {0,0,0,0,0,0,0,0};
      acc = wmma_f16(q0, wfrag(sR1, 0, j, 4, l), acc);
      acc = wmma_f16(q1, wfrag(sR1, 1, j, 4, l), acc);
      acc = bias_relu(acc, sBB1[(j << 4) + (l & 15)]);
      stage_put(stage, l, j, acc);
    }
  }
  asm volatile("" ::: "memory");
  {                                           // L3 64->64
    v16h q0 = stage_frag(stage, l, 0);
    v16h q1 = stage_frag(stage, l, 1);
#pragma unroll
    for (int j = 0; j < 4; ++j) {
      v8f acc = {0,0,0,0,0,0,0,0};
      acc = wmma_f16(q0, wfrag(sR2, 0, j, 4, l), acc);
      acc = wmma_f16(q1, wfrag(sR2, 1, j, 4, l), acc);
      acc = bias_relu(acc, sBB2[(j << 4) + (l & 15)]);
      stage_put(stage, l, j, acc);
    }
  }
  asm volatile("" ::: "memory");
  if (l < 16) {                               // L4 64->1 + ReLU
    float s = sB4;
#pragma unroll
    for (int k = 0; k < 64; ++k) s += (float)stage[l * STG + k] * sW4[k];
    out[base + l] = fmaxf(s, 0.0f);
  }
}

// ================= launch =================
// Input order: uids, iids, u_item_pad, i_user_pad, user_tbl, item_tbl, rate_tbl,
// then params pytree leaves with dict keys sorted alphabetically:
//   i_agg(b,w x2)=7..10, i_att(b,w x3)=11..16, i_g_u=17..22, rate_pred=23..30,
//   u_agg=31..34, u_att=35..40, u_g_v=41..46
extern "C" void kernel_launch(void* const* d_in, const int* in_sizes, int n_in,
                              void* d_out, int out_size, void* d_ws, size_t ws_size,
                              hipStream_t stream) {
  (void)in_sizes; (void)n_in; (void)out_size; (void)ws_size;
  const int*   uids     = (const int*)d_in[0];
  const int*   iids     = (const int*)d_in[1];
  const int*   u_pad    = (const int*)d_in[2];
  const int*   i_pad    = (const int*)d_in[3];
  const float* user_tbl = (const float*)d_in[4];
  const float* item_tbl = (const float*)d_in[5];
  const float* rate_tbl = (const float*)d_in[6];
  auto P = [&](int i) { return (const float*)d_in[i]; };

  float* ws    = (float*)d_ws;
  float* num_u = ws;                       // [B,64]
  float* den_u = num_u + BB * DD;          // [B]
  float* num_i = den_u + BB;               // [B,64]
  float* den_i = num_i + BB * DD;          // [B]
  float* h_i   = den_i + BB;               // [B,64]
  float* z_j   = h_i + BB * DD;            // [B,64]

  hipMemsetAsync(num_u, 0, (size_t)(2 * BB * DD + 2 * BB) * sizeof(float), stream);

  // 51200 tiles / (8 waves * TPW tiles) = 800 blocks
  const int branch_grid = (BB * LL) / (128 * TPW);
  graphrec_branch<<<branch_grid, 256, 0, stream>>>(
      uids, u_pad, user_tbl, item_tbl, rate_tbl,
      P(42), P(41), P(44), P(43), P(46), P(45),   // u_g_v  (w,b per layer)
      P(36), P(35), P(38), P(37), P(40), P(39),   // u_att
      num_u, den_u);
  graphrec_branch<<<branch_grid, 256, 0, stream>>>(
      iids, i_pad, item_tbl, user_tbl, rate_tbl,
      P(18), P(17), P(20), P(19), P(22), P(21),   // i_g_u
      P(12), P(11), P(14), P(13), P(16), P(15),   // i_att
      num_i, den_i);
  graphrec_agg<<<BB / 128, 256, 0, stream>>>(num_u, den_u, P(32), P(31), P(34), P(33), h_i);
  graphrec_agg<<<BB / 128, 256, 0, stream>>>(num_i, den_i, P(8),  P(7),  P(10), P(9),  z_j);
  graphrec_rate<<<BB / 128, 256, 0, stream>>>(h_i, z_j,
      P(24), P(23), P(26), P(25), P(28), P(27), P(30), P(29), (float*)d_out);
}
